// DimIxLoss_2379411882005
// MI455X (gfx1250) — compile-verified, matrix operand order checked by reading, not run on hardware
//
#include <hip/hip_runtime.h>
#include <hip/hip_fp16.h>

typedef __attribute__((ext_vector_type(16))) _Float16 v16h;
typedef __attribute__((ext_vector_type(8)))  _Float16 v8h;
typedef __attribute__((ext_vector_type(8)))  float    v8f;
typedef __attribute__((ext_vector_type(4)))  unsigned int u32x4;
typedef __attribute__((ext_vector_type(8)))  int i32x8;
typedef __attribute__((ext_vector_type(4)))  int i32x4;
typedef int vsi4 __attribute__((vector_size(16)));   // plain vector: trivially assignable across addrspaces

#define GASQ __attribute__((address_space(1)))
#define LASQ __attribute__((address_space(3)))

#ifndef __has_builtin
#define __has_builtin(x) 0
#endif

#if defined(__AMDGCN__)
#if __has_builtin(__builtin_amdgcn_global_load_async_to_lds_b128) && \
    __has_builtin(__builtin_amdgcn_s_wait_asynccnt)
#define HAVE_ASYNC_LDS 1
#endif
#if __has_builtin(__builtin_amdgcn_tensor_load_to_lds) && \
    __has_builtin(__builtin_amdgcn_s_wait_tensorcnt)
#define HAVE_TDM 1
#endif
#endif

constexpr int B = 32, F = 512, S = 1024;
constexpr int TOPK = S / 2;
constexpr float EPSN = 1e-10f;
constexpr int BS_STR = 40;   // padded LDS row stride in halves (80B: 16B-aligned, bank-conflict-free)
constexpr int AS_STR = 40;

// workspace layout (bytes)
constexpr size_t OFF_ACC  = 0;                                     // 3 floats
constexpr size_t OFF_MEAN = 1024;                                  // 3*B*F floats
constexpr size_t OFF_INV  = OFF_MEAN + sizeof(float) * 3 * B * F;  // 3*B*S floats
constexpr size_t OFF_HALF = OFF_INV  + sizeof(float) * 3 * B * S;  // 3 f16 tensors [B][S][F]
constexpr size_t HALF_ELEMS = (size_t)B * F * S;

__global__ void init_acc(float* acc) {
    if (threadIdx.x < 3) acc[threadIdx.x] = 0.0f;
}

// per-(tensor,b,f): mean over S
__global__ void mean_kernel(const float* __restrict__ x, const float* __restrict__ y,
                            const float* __restrict__ z, float* __restrict__ means) {
    int f = blockIdx.x, b = blockIdx.y, t = blockIdx.z;
    const float* in = (t == 0) ? x : ((t == 1) ? y : z);
    const GASQ float* row = (const GASQ float*)(in + (size_t)(b * F + f) * S);
    float s = 0.0f;
    for (int i = threadIdx.x; i < S; i += 256) s += row[i];
    __shared__ float red[256];
    red[threadIdx.x] = s;
    __syncthreads();
    for (int o = 128; o > 0; o >>= 1) {
        if (threadIdx.x < o) red[threadIdx.x] += red[threadIdx.x + o];
        __syncthreads();
    }
    if (threadIdx.x == 0) means[(t * B + b) * F + f] = red[0] * (1.0f / (float)S);
}

// per-(tensor,b,s): 1/(||centered column||_2 + eps)
__global__ void invnorm_kernel(const float* __restrict__ x, const float* __restrict__ y,
                               const float* __restrict__ z, const float* __restrict__ means,
                               float* __restrict__ invn) {
    int t = blockIdx.z, b = blockIdx.y;
    int s = blockIdx.x * 256 + threadIdx.x;
    const float* in = (t == 0) ? x : ((t == 1) ? y : z);
    const GASQ float* mrow = (const GASQ float*)(means + (t * B + b) * F);
    const GASQ float* base = (const GASQ float*)(in + (size_t)b * F * S);
    float ss = 0.0f;
    for (int f = 0; f < F; ++f) {
        float v = base[(size_t)f * S + s] - mrow[f];
        ss += v * v;
    }
    invn[(t * B + b) * S + s] = 1.0f / (sqrtf(ss) + EPSN);
}

// center+normalize+convert+transpose: out[b][s][f] = (in[b][f][s]-mean)*inv, f16, K-contiguous
__global__ void transconv_kernel(const float* __restrict__ x, const float* __restrict__ y,
                                 const float* __restrict__ z, const float* __restrict__ means,
                                 const float* __restrict__ invn, _Float16* __restrict__ half_base) {
    __shared__ _Float16 tile[32][33];
    int t = blockIdx.z, b = blockIdx.y;
    int ftile = blockIdx.x % (F / 32), stile = blockIdx.x / (F / 32);
    int tf = ftile * 32, ts = stile * 32;
    int tx = threadIdx.x, ty = threadIdx.y;  // block (32,8)
    const float* in = (t == 0) ? x : ((t == 1) ? y : z);
    const GASQ float* mrow = (const GASQ float*)(means + (t * B + b) * F);
    const GASQ float* irow = (const GASQ float*)(invn + (t * B + b) * S);
    const GASQ float* base = (const GASQ float*)(in + (size_t)b * F * S);
    float inv = irow[ts + tx];
    for (int i = 0; i < 4; ++i) {
        int f = tf + i * 8 + ty;
        float v = (base[(size_t)f * S + ts + tx] - mrow[f]) * inv;
        tile[i * 8 + ty][tx] = (_Float16)v;
    }
    __syncthreads();
    GASQ _Float16* ob = (GASQ _Float16*)(half_base + (size_t)t * HALF_ELEMS + (size_t)b * S * F);
    for (int i = 0; i < 4; ++i) {
        int srow = ts + i * 8 + ty;
        ob[(size_t)srow * F + tf + tx] = tile[tx][i * 8 + ty];
    }
}

#if defined(HAVE_TDM)
// One TDM descriptor moves the whole 1024x32-half B panel into LDS, reproducing the
// 40-half padded row stride via pad_enable (interval=16 DWORDs -> code 3, amount=4 DWORDs -> code 3).
// D# layout per CDNA5 ISA.
__device__ inline void tdm_load_panel(const GASQ _Float16* src, unsigned lds_byte_addr) {
    unsigned long long ga = (unsigned long long)(size_t)src;
    u32x4 g0;
    g0[0] = 1u;                                                   // count=1, user descriptor
    g0[1] = lds_byte_addr;                                        // lds_addr
    g0[2] = (unsigned)(ga & 0xFFFFFFFFu);                         // global_addr[31:0]
    g0[3] = (unsigned)((ga >> 32) & 0x01FFFFFFu) | (2u << 30);    // global_addr[56:32] | type=2
    i32x8 g1;
    g1[0] = (1 << 16) | (1 << 20) | (3 << 22) | (3 << 25);  // data_size=2B, pad_en, intv=16dw, amt=4dw
    g1[1] = (int)(((unsigned)F & 0xFFFFu) << 16);           // tensor_dim0[15:0] = 512
    g1[2] = (int)(((unsigned)S & 0xFFFFu) << 16);           // dim0[31:16]=0 | tensor_dim1[15:0]=1024
    g1[3] = (int)(32u << 16);                               // dim1[31:16]=0 | tile_dim0=32
    g1[4] = (int)(unsigned)S;                               // tile_dim1=1024 | tile_dim2=0
    g1[5] = (int)F;                                         // tensor_dim0_stride[31:0] = 512
    g1[6] = 0;                                              // stride[47:32]=0 | dim1_stride[15:0]=0
    g1[7] = 0;
    i32x4 gz = {0, 0, 0, 0};
#if __clang_major__ >= 23
    i32x8 gz8 = {0, 0, 0, 0, 0, 0, 0, 0};
    __builtin_amdgcn_tensor_load_to_lds(g0, g1, gz, gz, gz8, 0);
#else
    __builtin_amdgcn_tensor_load_to_lds(g0, g1, gz, gz, 0);
#endif
}
#endif

// one 16-row strip-GEMM from [S][F] (K-contiguous) operands:
// Gs[m][n] (=/+=) sum_k AT[b][s0+m][k] * BT[b][n][k]
__device__ inline void gemm_strip(const _Float16* __restrict__ AT,
                                  const _Float16* __restrict__ BT,
                                  int b, int s0, float* Gs, _Float16* As, _Float16* Bs,
                                  bool add) {
    const int tid = threadIdx.x;
    const int lane = tid & 31, wave = tid >> 5;
    const int m = lane & 15, kb = (lane >> 4) * 8;
    const GASQ _Float16* Abase = (const GASQ _Float16*)(AT + (size_t)b * S * F);
    const GASQ _Float16* Bbase = (const GASQ _Float16*)(BT + (size_t)b * S * F);
#if defined(HAVE_TDM)
    const unsigned bs_lds = (unsigned)(unsigned long long)(LASQ _Float16*)Bs;
#endif

    v8f cacc[4];
    for (int j = 0; j < 4; ++j)
        for (int r = 0; r < 8; ++r) cacc[j][r] = 0.0f;

    for (int k0 = 0; k0 < F; k0 += 32) {
        __syncthreads();  // previous consumers done
        // ---- A strip: As[mm][0..31] <- AT[s0+mm][k0..k0+31] ----
        if (tid < 64) {
            int mm = tid >> 2, ch = tid & 3;
            const GASQ _Float16* srcA = Abase + (size_t)(s0 + mm) * F + k0 + ch * 8;
            _Float16* dstA = As + mm * AS_STR + ch * 8;
#if defined(HAVE_ASYNC_LDS)
            __builtin_amdgcn_global_load_async_to_lds_b128(
                (GASQ vsi4*)srcA, (LASQ vsi4*)dstA, 0, 0);
#else
            *(vsi4*)dstA = *(const GASQ vsi4*)srcA;
#endif
        }
        // ---- B panel: Bs[n][0..31] <- BT[n][k0..k0+31] ----
#if defined(HAVE_TDM)
        if (tid == 0) tdm_load_panel(Bbase + k0, bs_lds);
#elif defined(HAVE_ASYNC_LDS)
        for (int it = 0; it < 8; ++it) {
            int e = it * 512 + tid;
            int n = e >> 2, ch = e & 3;
            __builtin_amdgcn_global_load_async_to_lds_b128(
                (GASQ vsi4*)(Bbase + (size_t)n * F + k0 + ch * 8),
                (LASQ vsi4*)(Bs + n * BS_STR + ch * 8), 0, 0);
        }
#else
        for (int it = 0; it < 8; ++it) {
            int e = it * 512 + tid;
            int n = e >> 2, ch = e & 3;
            *(vsi4*)(Bs + n * BS_STR + ch * 8) =
                *(const GASQ vsi4*)(Bbase + (size_t)n * F + k0 + ch * 8);
        }
        if (k0 + 32 < F)
            __builtin_prefetch((const void*)(BT + (size_t)b * S * F + (size_t)(tid * 2) * F + k0 + 32), 0, 1);
#endif
#if defined(HAVE_TDM)
        __builtin_amdgcn_s_wait_tensorcnt(0);
#endif
#if defined(HAVE_ASYNC_LDS)
        __builtin_amdgcn_s_wait_asynccnt(0);
#endif
        __syncthreads();

        // A fragment: two contiguous b128 runs per lane
        union { v16h v; struct { v8h lo, hi; } s; } ua;
        ua.s.lo = *(const v8h*)(As + m * AS_STR + kb);
        ua.s.hi = *(const v8h*)(As + m * AS_STR + 16 + kb);
        for (int j = 0; j < 4; ++j) {
            int n = (wave * 4 + j) * 16 + m;
            union { v16h v; struct { v8h lo, hi; } s; } ub;
            ub.s.lo = *(const v8h*)(Bs + n * BS_STR + kb);
            ub.s.hi = *(const v8h*)(Bs + n * BS_STR + 16 + kb);
            cacc[j] = __builtin_amdgcn_wmma_f32_16x16x32_f16(
                false, ua.v, false, ub.v, (short)0, cacc[j], false, false);
        }
    }
    __syncthreads();
    // C layout: lanes 0-15 rows 0-7, lanes 16-31 rows 8-15; N = lane&15
    for (int j = 0; j < 4; ++j) {
        int col = (wave * 4 + j) * 16 + m;
        int rowbase = (lane >> 4) * 8;
        for (int r = 0; r < 8; ++r) {
            float v = cacc[j][r];
            int idx = (rowbase + r) * 1024 + col;
            if (add) Gs[idx] += v; else Gs[idx] = v;
        }
    }
}

__launch_bounds__(512)
__global__ void pair_kernel(const _Float16* __restrict__ half_base, float* __restrict__ acc) {
    __shared__ __align__(16) _Float16 As[16 * AS_STR];
    __shared__ __align__(16) _Float16 Bs[1024 * BS_STR];
    __shared__ float Gs[16 * 1024];

    int s0 = blockIdx.x * 16, b = blockIdx.y, pr = blockIdx.z;
    const _Float16* T0 = half_base;
    const _Float16* T1 = half_base + HALF_ELEMS;
    const _Float16* T2 = half_base + 2 * HALF_ELEMS;
    // pairs: (X,Y) = (x,y), (y,z), (z,x);  G = Y^T X, Gsym = G + G^T
    const _Float16* X = (pr == 0) ? T0 : ((pr == 1) ? T1 : T2);
    const _Float16* Y = (pr == 0) ? T1 : ((pr == 1) ? T2 : T0);

    gemm_strip(Y, X, b, s0, Gs, As, Bs, false);  // (Y_strip)^T X
    gemm_strip(X, Y, b, s0, Gs, As, Bs, true);   // + (X_strip)^T Y
    __syncthreads();

    // one wave32 per row: softmax + exact top-k(512) + weighted index sum
    int lane = threadIdx.x & 31, w = threadIdx.x >> 5;
    const float* row = Gs + w * 1024;
    float pv[32];
    float gmax = -1e30f;
    for (int i = 0; i < 32; ++i) {
        float g = row[lane + 32 * i];
        pv[i] = g;
        gmax = fmaxf(gmax, g);
    }
    for (int o = 16; o; o >>= 1) gmax = fmaxf(gmax, __shfl_xor(gmax, o, 32));
    float psum = 0.0f;
    for (int i = 0; i < 32; ++i) {
        float p = __expf(pv[i] - gmax);  // top-k invariant under 1/psum scaling
        pv[i] = p;
        psum += p;
    }
    for (int o = 16; o; o >>= 1) psum += __shfl_xor(psum, o, 32);

    // binary search on float bit patterns (order-preserving for positives):
    // largest lo with count(bits >= lo) >= TOPK  ==>  lo == k-th largest value's bits
    unsigned lo = 0u, hi = 0x3F800001u;  // p in (0, 1]
    while (hi - lo > 1u) {
        unsigned mid = (lo + hi) >> 1;
        int cnt = 0;
        for (int i = 0; i < 32; ++i) cnt += (__float_as_uint(pv[i]) >= mid) ? 1 : 0;
        for (int o = 16; o; o >>= 1) cnt += __shfl_xor(cnt, o, 32);
        if (cnt >= TOPK) lo = mid; else hi = mid;
    }

    int srow = s0 + w;
    float sgt = 0.0f, seq = 0.0f;
    int cgt = 0, ceq = 0;
    for (int i = 0; i < 32; ++i) {
        unsigned bits = __float_as_uint(pv[i]);
        float d = (float)(lane + 32 * i - srow);
        if (bits > lo)       { sgt += pv[i] * d; ++cgt; }
        else if (bits == lo) { seq += pv[i] * d; ++ceq; }
    }
    for (int o = 16; o; o >>= 1) {
        sgt += __shfl_xor(sgt, o, 32);
        seq += __shfl_xor(seq, o, 32);
        cgt += __shfl_xor(cgt, o, 32);
        ceq += __shfl_xor(ceq, o, 32);
    }
    if (lane == 0) {
        float stop = sgt + ((ceq > 0) ? (float)(TOPK - cgt) * (seq / (float)ceq) : 0.0f);
        float crow = stop / (psum * (float)TOPK * (float)S);   // mean over k, /S, softmax scale
        atomicAdd(acc + pr, crow * (1.0f / (float)(B * S)));   // mean over [B,S]
    }
}

__global__ void final_kernel(const float* __restrict__ acc, float* __restrict__ out) {
    float a0 = acc[0], a1 = acc[1], a2 = acc[2];
    // faithful to reference: exp(-m + min(m) - 0.1) with scalar m
    float r = __expf(-a0 + a0 - 0.1f) + __expf(-a1 + a1 - 0.1f) + __expf(-a2 + a2 - 0.1f);
    out[0] = r;
}

extern "C" void kernel_launch(void* const* d_in, const int* in_sizes, int n_in,
                              void* d_out, int out_size, void* d_ws, size_t ws_size,
                              hipStream_t stream) {
    const float* x = (const float*)d_in[0];
    const float* y = (const float*)d_in[1];
    const float* z = (const float*)d_in[2];
    float* out = (float*)d_out;
    char* ws = (char*)d_ws;
    float* acc = (float*)(ws + OFF_ACC);
    float* means = (float*)(ws + OFF_MEAN);
    float* invn = (float*)(ws + OFF_INV);
    _Float16* halves = (_Float16*)(ws + OFF_HALF);

    init_acc<<<1, 64, 0, stream>>>(acc);
    mean_kernel<<<dim3(F, B, 3), 256, 0, stream>>>(x, y, z, means);
    invnorm_kernel<<<dim3(S / 256, B, 3), 256, 0, stream>>>(x, y, z, means, invn);
    transconv_kernel<<<dim3((F / 32) * (S / 32), B, 3), dim3(32, 8), 0, stream>>>(
        x, y, z, means, invn, halves);
    pair_kernel<<<dim3(S / 16, B, 3), 512, 0, stream>>>(halves, acc);
    final_kernel<<<1, 1, 0, stream>>>(acc, out);
}
